// Attri_classify_net_22_71107478552581
// MI455X (gfx1250) — matrix operand order repaired and checked by reading, not compiled, not run on hardware
//
#include <hip/hip_runtime.h>

typedef __attribute__((ext_vector_type(16))) _Float16 v16h;
typedef __attribute__((ext_vector_type(8)))  float    v8f;

#define FDIM   256
#define BATCH  131072
#define NPAD   96
#define NCLS   94
#define NHEADS 21
#define BN_EPS 1e-5f

// head index for each global class column (94 valid + 2 pad)
__constant__ int c_head_of[NPAD] = {
  0,0,0,0,0,
  1,1,1,1,
  2,2,2,2,2,2,
  3,3,3,
  4,4,4,4,4,4,
  5,5,5,5,
  6,6,6,
  7,7,7,7,
  8,8,8,8,8,
  9,9,9,9,
  10,10,10,
  11,11,11,11,11,11,
  12,12,12,12,12,12,12,
  13,13,13,13,13,
  14,14,14,
  15,15,15,15,15,15,15,
  16,16,16,16,16,
  17,17,17,
  18,18,18,18,
  19,19,19,
  20,20,20,20,
  0,0 };
__constant__ int c_head_off[NHEADS + 1] = {
  0,5,9,15,18,24,28,31,35,40,44,47,53,60,65,68,75,80,83,87,90,94 };

struct HeadPtrs {
  const float* W[NHEADS];
  const float* g[NHEADS];
  const float* bt[NHEADS];
};

// ---------------- zero accumulators ----------------
__global__ void zero_kernel(float* sums, float* sumsq, int* counts, float* accum) {
  int t = threadIdx.x;
  if (t < NPAD)   { sums[t] = 0.f; sumsq[t] = 0.f; }
  if (t < 32)     counts[t] = 0;
  if (t == 0)     accum[0] = 0.f;
}

// ---------------- pack W (f32 -> f16, padded [96][256]) + gamma/beta ----------------
__global__ void pack_kernel(HeadPtrs hp, _Float16* __restrict__ Wh,
                            float* __restrict__ gpack, float* __restrict__ bpack) {
  int r = blockIdx.x;
  if (r < NCLS) {
    int h   = c_head_of[r];
    int loc = r - c_head_off[h];
    const float* src = hp.W[h] + (size_t)loc * FDIM;
    for (int k = threadIdx.x; k < FDIM; k += blockDim.x)
      Wh[(size_t)r * FDIM + k] = (_Float16)src[k];
    if (threadIdx.x == 0) { gpack[r] = hp.g[h][loc]; bpack[r] = hp.bt[h][loc]; }
  } else {
    for (int k = threadIdx.x; k < FDIM; k += blockDim.x)
      Wh[(size_t)r * FDIM + k] = (_Float16)0.f;
    if (threadIdx.x == 0) { gpack[r] = 1.f; bpack[r] = 0.f; }
  }
}

// ---------------- z = x @ W^T via V_WMMA_F32_16X16X32_F16 ----------------
// block = 256 threads = 8 waves; each wave owns a 16-row M-tile and all 6 N-tiles.
__global__ __launch_bounds__(256) void gemm_wmma(const float* __restrict__ x,
                                                 const _Float16* __restrict__ Wh,
                                                 float* __restrict__ z) {
  const int lane = threadIdx.x & 31;
  const int wave = threadIdx.x >> 5;
  const int l16  = lane & 15;
  const int hi   = lane >> 4;                 // 0 or 1 (lane half)
  const int rowBase = (blockIdx.x * 8 + wave) * 16;
  const float* xr = x + (size_t)(rowBase + l16) * FDIM;

  v8f acc[6];
  const v8f z8 = {0.f, 0.f, 0.f, 0.f, 0.f, 0.f, 0.f, 0.f};
#pragma unroll
  for (int n = 0; n < 6; ++n) acc[n] = z8;

#pragma unroll
  for (int kb = 0; kb < 8; ++kb) {
    const int k0 = kb * 32;
    const int ka = k0 + hi * 16;              // A: 16 consecutive K values for this lane half
    float4 p0 = *(const float4*)(xr + ka);
    float4 p1 = *(const float4*)(xr + ka + 4);
    float4 p2 = *(const float4*)(xr + ka + 8);
    float4 p3 = *(const float4*)(xr + ka + 12);
    v16h a;
    a[0]=(_Float16)p0.x;  a[1]=(_Float16)p0.y;  a[2]=(_Float16)p0.z;  a[3]=(_Float16)p0.w;
    a[4]=(_Float16)p1.x;  a[5]=(_Float16)p1.y;  a[6]=(_Float16)p1.z;  a[7]=(_Float16)p1.w;
    a[8]=(_Float16)p2.x;  a[9]=(_Float16)p2.y;  a[10]=(_Float16)p2.z; a[11]=(_Float16)p2.w;
    a[12]=(_Float16)p3.x; a[13]=(_Float16)p3.y; a[14]=(_Float16)p3.z; a[15]=(_Float16)p3.w;

    // B fragment: lane holds 16 consecutive K of column (ntile*16 + l16)
    const _Float16* wb = Wh + (size_t)l16 * FDIM + k0 + hi * 16;
#pragma unroll
    for (int n = 0; n < 6; ++n) {
      v16h b = *(const v16h*)(wb + (size_t)n * 16 * FDIM);
      acc[n] = __builtin_amdgcn_wmma_f32_16x16x32_f16(
          false, a, false, b, (short)0, acc[n], false, false);
    }
  }

  float* zr = z + (size_t)rowBase * NPAD;
#pragma unroll
  for (int n = 0; n < 6; ++n)
#pragma unroll
    for (int v = 0; v < 8; ++v)
      zr[(size_t)(v + 8 * hi) * NPAD + n * 16 + l16] = acc[n][v];
}

// ---------------- per-head row counts ----------------
__global__ __launch_bounds__(256) void counts_kernel(const int* __restrict__ sec,
                                                     int* __restrict__ counts) {
  __shared__ int s_cnt[NHEADS];
  if (threadIdx.x < NHEADS) s_cnt[threadIdx.x] = 0;
  __syncthreads();
  int row = blockIdx.x * 256 + threadIdx.x;
  atomicAdd(&s_cnt[sec[row]], 1);
  __syncthreads();
  if (threadIdx.x < NHEADS) atomicAdd(&counts[threadIdx.x], s_cnt[threadIdx.x]);
}

// ---------------- masked sum / sumsq per class column ----------------
__global__ __launch_bounds__(256) void stats_kernel(const float* __restrict__ z,
                                                    const int* __restrict__ sec,
                                                    float* __restrict__ sums,
                                                    float* __restrict__ sumsq) {
  __shared__ float s_sum[NPAD], s_sq[NPAD];
  int tid = threadIdx.x;
  if (tid < NPAD) { s_sum[tid] = 0.f; s_sq[tid] = 0.f; }
  __syncthreads();
  const int base = blockIdx.x * (256 * 12);
#pragma unroll
  for (int i = 0; i < 12; ++i) {
    int idx = base + i * 256 + tid;          // < 131072*96, fits int32
    int row = idx / NPAD;
    int col = idx - row * NPAD;
    if (col < NCLS && sec[row] == c_head_of[col]) {
      float v = z[idx];
      atomicAdd(&s_sum[col], v);
      atomicAdd(&s_sq[col], v * v);
    }
  }
  __syncthreads();
  if (tid < NCLS) {
    atomicAdd(&sums[tid],  s_sum[tid]);
    atomicAdd(&sumsq[tid], s_sq[tid]);
  }
}

// ---------------- fold BN stats + gamma/beta into per-class affine ----------------
__global__ void meanvar_kernel(const float* sums, const float* sumsq, const int* counts,
                               const float* gpack, const float* bpack,
                               float* a_arr, float* b_arr, float* invn) {
  int c = threadIdx.x;
  if (c < NCLS) {
    int h  = c_head_of[c];
    int cn = counts[h] > 1 ? counts[h] : 1;
    float n   = (float)cn;
    float m   = sums[c] / n;
    float var = sumsq[c] / n - m * m;
    var = fmaxf(var, 0.f);
    float aa = gpack[c] * rsqrtf(var + BN_EPS);
    a_arr[c] = aa;
    b_arr[c] = bpack[c] - m * aa;
  }
  if (c < NHEADS) {
    int cn = counts[c] > 1 ? counts[c] : 1;
    invn[c] = 1.f / (float)cn;
  }
}

// ---------------- fused normalize + log-softmax + NLL, weighted by 1/n_head ----------------
__global__ __launch_bounds__(256) void loss_kernel(const float* __restrict__ z,
                                                   const int* __restrict__ sec,
                                                   const int* __restrict__ lab,
                                                   const float* __restrict__ a_arr,
                                                   const float* __restrict__ b_arr,
                                                   const float* __restrict__ invn,
                                                   float* __restrict__ accum) {
  __shared__ float sa[NCLS], sb[NCLS], sn[NHEADS];
  __shared__ float s_acc;
  int tid = threadIdx.x;
  if (tid < NCLS)   { sa[tid] = a_arr[tid]; sb[tid] = b_arr[tid]; }
  if (tid < NHEADS) sn[tid] = invn[tid];
  if (tid == 0)     s_acc = 0.f;
  __syncthreads();

  int row = blockIdx.x * 256 + tid;
  int h   = sec[row];
  int l   = lab[row];
  int off = c_head_off[h];
  int C   = c_head_off[h + 1] - off;
  const float* zr = z + (size_t)row * NPAD + off;

  float zn[8];
  float mx = -1e30f;
  for (int j = 0; j < C; ++j) {
    float t = sa[off + j] * zr[j] + sb[off + j];
    zn[j] = t;
    mx = fmaxf(mx, t);
  }
  float se = 0.f;
  for (int j = 0; j < C; ++j) se += __expf(zn[j] - mx);
  float nll = (mx + __logf(se)) - zn[l];

  atomicAdd(&s_acc, nll * sn[h]);
  __syncthreads();
  if (tid == 0) atomicAdd(accum, s_acc);
}

__global__ void finalize_kernel(const float* accum, float* out) {
  out[0] = accum[0] * (1.0f / 21.0f);
}

// ---------------- launch ----------------
extern "C" void kernel_launch(void* const* d_in, const int* in_sizes, int n_in,
                              void* d_out, int out_size, void* d_ws, size_t ws_size,
                              hipStream_t stream) {
  static const int head_off_h[NHEADS + 1] = {
    0,5,9,15,18,24,28,31,35,40,44,47,53,60,65,68,75,80,83,87,90,94 };

  const float* x = (const float*)d_in[0];
  HeadPtrs hp;
  const int* sec;
  const int* att;
  if (n_in >= 66) {                       // tuples flattened into separate inputs
    for (int i = 0; i < NHEADS; ++i) {
      hp.W[i]  = (const float*)d_in[1 + i];
      hp.g[i]  = (const float*)d_in[22 + i];
      hp.bt[i] = (const float*)d_in[43 + i];
    }
    sec = (const int*)d_in[64];
    att = (const int*)d_in[65];
  } else {                                // tuples concatenated into single buffers
    const float* Wall = (const float*)d_in[1];
    const float* gall = (const float*)d_in[2];
    const float* ball = (const float*)d_in[3];
    for (int i = 0; i < NHEADS; ++i) {
      hp.W[i]  = Wall + (size_t)head_off_h[i] * FDIM;
      hp.g[i]  = gall + head_off_h[i];
      hp.bt[i] = ball + head_off_h[i];
    }
    sec = (const int*)d_in[4];
    att = (const int*)d_in[5];
  }

  // workspace layout (~50.4 MB)
  char* ws = (char*)d_ws;
  size_t off = 0;
  float*     z      = (float*)(ws + off);  off += (size_t)BATCH * NPAD * 4;   // 50331648
  _Float16*  Wh     = (_Float16*)(ws + off); off += (size_t)NPAD * FDIM * 2;  // 49152
  float*     gpack  = (float*)(ws + off);  off += NPAD * 4;
  float*     bpack  = (float*)(ws + off);  off += NPAD * 4;
  float*     sums   = (float*)(ws + off);  off += NPAD * 4;
  float*     sumsq  = (float*)(ws + off);  off += NPAD * 4;
  float*     a_arr  = (float*)(ws + off);  off += NPAD * 4;
  float*     b_arr  = (float*)(ws + off);  off += NPAD * 4;
  int*       counts = (int*)(ws + off);    off += 32 * 4;
  float*     invn   = (float*)(ws + off);  off += 32 * 4;
  float*     accum  = (float*)(ws + off);  off += 4;

  zero_kernel<<<1, 256, 0, stream>>>(sums, sumsq, counts, accum);
  pack_kernel<<<NPAD, 256, 0, stream>>>(hp, Wh, gpack, bpack);
  gemm_wmma<<<BATCH / 128, 256, 0, stream>>>(x, Wh, z);
  counts_kernel<<<BATCH / 256, 256, 0, stream>>>(sec, counts);
  stats_kernel<<<(BATCH * NPAD) / (256 * 12), 256, 0, stream>>>(z, sec, sums, sumsq);
  meanvar_kernel<<<1, 128, 0, stream>>>(sums, sumsq, counts, gpack, bpack, a_arr, b_arr, invn);
  loss_kernel<<<BATCH / 256, 256, 0, stream>>>(z, sec, att, a_arr, b_arr, invn, accum);
  finalize_kernel<<<1, 1, 0, stream>>>(accum, (float*)d_out);
}
